// SpatialAxialAttention_18622978196124
// MI455X (gfx1250) — matrix operand order repaired and checked by reading, not compiled, not run on hardware
//
#include <hip/hip_runtime.h>
#include <math.h>

// ---------------------------------------------------------------------------
// Types for CDNA5 WMMA (wave32): 16x16x32 bf16 -> f32
// ---------------------------------------------------------------------------
typedef __attribute__((ext_vector_type(16))) __bf16 v16bf;
typedef __attribute__((ext_vector_type(8)))  float  v8f;
typedef __attribute__((ext_vector_type(4)))  unsigned int v4u;
typedef __attribute__((ext_vector_type(8)))  int  v8i;
typedef __attribute__((ext_vector_type(4)))  int  v4i;

union Frag16 { uint4 q[2]; v16bf f; };

static __device__ __forceinline__ unsigned short f32_to_bf16_rne(float f) {
  unsigned u = __float_as_uint(f);
  u += 0x7FFFu + ((u >> 16) & 1u);   // round-to-nearest-even
  return (unsigned short)(u >> 16);
}

// ---------------------------------------------------------------------------
// Problem constants: B*T=2, HEADS=16, DIM_HEAD=64, S=256, dim=1024, top-k=144.
// ---------------------------------------------------------------------------
constexpr int BT = 2, HEADS = 16, S = 256, DHEAD = 64, TOPK = 144;

// GEMM tiling: workgroup = 64(M) x 128(N), 8 waves each doing 16x64.
constexpr int BM = 64, BN = 128, BK = 32;
constexpr int LDA = 40;  // bf16 elements; 80B row stride breaks bank aliasing
constexpr int LDB = 40;

// ---------------------------------------------------------------------------
// f32 -> bf16 conversion kernel
// ---------------------------------------------------------------------------
__global__ void f32_to_bf16_kernel(const float* __restrict__ in,
                                   unsigned short* __restrict__ out, int n) {
  int i = blockIdx.x * blockDim.x + threadIdx.x;
  if (i < n) out[i] = f32_to_bf16_rne(in[i]);
}

// ---------------------------------------------------------------------------
// WMMA GEMM:  C[m][n] = sum_k A[m][k] * B[n][k]  (torch Linear weight layout).
// Register-staged double buffering: tile k+1 is loaded into VGPRs while the
// WMMAs for tile k run, so the loadcnt wait lands at the next ds_store.
// MODE 0: qkv projection epilogue -> scatter into q/k/v (bt,head,s,d) buffers.
// MODE 1: output projection epilogue -> add bias, write f32 row-major.
// ---------------------------------------------------------------------------
template <int KDIM, int MODE>
__global__ __launch_bounds__(256) void gemm_bf16_wmma(
    const unsigned short* __restrict__ A,   // [M][KDIM] bf16 bits
    const unsigned short* __restrict__ B,   // [N][KDIM] bf16 bits
    const float* __restrict__ bias,         // MODE 1 only
    float* __restrict__ out0,               // MODE0: q   MODE1: final out
    float* __restrict__ out1,               // MODE0: k
    float* __restrict__ out2)               // MODE0: v
{
  __shared__ unsigned short sA[BM * LDA];
  __shared__ unsigned short sB[BN * LDB];

  const int t    = threadIdx.x;
  const int m0   = blockIdx.x * BM;
  const int n0   = blockIdx.y * BN;
  const int lane = t & 31, wave = t >> 5;
  const int half = lane >> 4, mr = lane & 15;
  const int waveM = (wave & 3) * 16;
  const int waveN = (wave >> 2) * 64;

  v8f acc[4];
#pragma unroll
  for (int j = 0; j < 4; ++j)
#pragma unroll
    for (int e = 0; e < 8; ++e) acc[j][e] = 0.f;

  // cooperative-load assignments (16B vector moves)
  const int arow = t >> 2, ac = (t & 3) * 8;    // 64 rows x 4 chunks of 8 bf16
  const int brow = t >> 1, bc = (t & 1) * 16;   // 128 rows x 2 chunks of 16 bf16
  const unsigned short* aptr = &A[(size_t)(m0 + arow) * KDIM + ac];
  const unsigned short* bptr = &B[(size_t)(n0 + brow) * KDIM + bc];

  // stage tile 0 in registers
  uint4 ra  = *(const uint4*)aptr;
  uint4 rb0 = ((const uint4*)bptr)[0];
  uint4 rb1 = ((const uint4*)bptr)[1];

  for (int k0 = 0; k0 < KDIM; k0 += BK) {
    *(uint4*)&sA[arow * LDA + ac] = ra;
    uint4* bdst = (uint4*)&sB[brow * LDB + bc];
    bdst[0] = rb0;
    bdst[1] = rb1;
    __syncthreads();

    // issue next tile's global loads; wait is deferred to next ds_store
    if (k0 + BK < KDIM) {
      ra  = *(const uint4*)(aptr + k0 + BK);
      rb0 = ((const uint4*)(bptr + k0 + BK))[0];
      rb1 = ((const uint4*)(bptr + k0 + BK))[1];
    }

    // A fragment: lane row M = lane%16, K-half = lane/16, e -> K contig
    // (matches B): lanes 0-15: e0-15=K0-15; lanes 16-31: e0-15=K16-31.
    Frag16 af, bfr[4];
    af.q[0] = *(const uint4*)&sA[(waveM + mr) * LDA + half * 16];
    af.q[1] = *(const uint4*)&sA[(waveM + mr) * LDA + half * 16 + 8];
#pragma unroll
    for (int j = 0; j < 4; ++j) {
      // B fragment: lane column N = lane%16, K-half = lane/16, e -> K contig.
      const uint4* pb =
          (const uint4*)&sB[(waveN + j * 16 + mr) * LDB + half * 16];
      bfr[j].q[0] = pb[0];
      bfr[j].q[1] = pb[1];
    }
#pragma unroll
    for (int j = 0; j < 4; ++j) {
      acc[j] = __builtin_amdgcn_wmma_f32_16x16x32_bf16(
          false, af.f, false, bfr[j].f, (short)0, acc[j], false, false);
    }
    __syncthreads();
  }

  // C/D layout: lane L, VGPR r -> M = r + (L/16)*8, N = L%16.
#pragma unroll
  for (int j = 0; j < 4; ++j) {
    const int n = n0 + waveN + j * 16 + mr;
#pragma unroll
    for (int r = 0; r < 8; ++r) {
      const int m = m0 + waveM + half * 8 + r;
      const float val = acc[j][r];
      if (MODE == 0) {
        const int part = n >> 10, rem = n & 1023;
        const int head = rem >> 6, d = rem & 63;
        const int bt = m >> 8, s = m & 255;
        float* dst = (part == 0) ? out0 : ((part == 1) ? out1 : out2);
        dst[(((size_t)(bt * HEADS + head) * S) + s) * DHEAD + d] = val;
      } else {
        out0[(size_t)m * 1024 + n] = val + bias[n];
      }
    }
  }
}

// ---------------------------------------------------------------------------
// Axial RoPE (freqs_for='pixel'), in place on q and k.
// ---------------------------------------------------------------------------
__global__ void rope_kernel(float* __restrict__ q, float* __restrict__ k) {
  const int i = blockIdx.x * blockDim.x + threadIdx.x;  // 524288 total
  const int pair  = i & 31;
  const int s     = (i >> 5) & 255;
  const int head  = (i >> 13) & 15;
  const int bt    = (i >> 17) & 1;
  const int which = (i >> 18) & 1;
  float* t = (which ? k : q) +
             (((size_t)(bt * HEADS + head) * S + s) * DHEAD) + pair * 2;
  const int hp = s >> 4, wp = s & 15;
  const float pos  = (pair < 16) ? (-1.f + hp * (2.f / 15.f))
                                 : (-1.f + wp * (2.f / 15.f));
  const int bi     = pair & 15;
  const float base = (1.f + bi * (127.f / 15.f)) * 3.14159265358979323846f;
  const float f    = pos * base;
  const float c = cosf(f), sn = sinf(f);
  const float e = t[0], o = t[1];
  t[0] = e * c - o * sn;
  t[1] = o * c + e * sn;
}

// ---------------------------------------------------------------------------
// Tensor Data Mover staging (CDNA5 ISA).
// Builds a 2D-tile D# (groups 0..3) and issues TENSOR_LOAD_TO_LDS. The K slab
// uses TDM LDS padding (pad_interval=64 DW -> code 5, pad_amount=2 DW -> code
// 1) to produce the stride-66 bank-conflict-free layout in hardware.
// This toolchain exposes the 6-arg builtin form:
//   (uint32x4 g0, int32x8 g1, int32x4 g2, int32x4 g3, int32x8 extra, i32 cpol)
// ---------------------------------------------------------------------------
#if defined(__HIP_DEVICE_COMPILE__) &&                                        \
    __has_builtin(__builtin_amdgcn_tensor_load_to_lds) &&                     \
    __has_builtin(__builtin_amdgcn_s_wait_tensorcnt)
#define ATTN_USE_TDM 1
#else
#define ATTN_USE_TDM 0
#endif

#if ATTN_USE_TDM
static __device__ __forceinline__ void tdm_load_2d_f32(
    const float* gaddr, unsigned lds_byte_off,
    unsigned tile_d0, unsigned tile_d1,      // tile size, f32 elements
    unsigned tensor_d0, unsigned tensor_d1,  // tensor size, f32 elements
    unsigned d0_stride,                      // row stride, f32 elements
    bool pad_en, unsigned pad_interval_code, unsigned pad_amount_code) {
  const unsigned long long ga = (unsigned long long)(size_t)gaddr;
  v4u g0;
  g0[0] = 1u;                                  // count=1, user-mode descriptor
  g0[1] = lds_byte_off;                        // lds_addr [63:32]
  g0[2] = (unsigned)(ga & 0xFFFFFFFFu);        // global_addr[31:0]
  g0[3] = (unsigned)((ga >> 32) & 0x01FFFFFFu) // global_addr[56:32]
          | (2u << 30);                        // type=2 ("image")
  v8i g1;
  unsigned w0 = (2u << 16);                    // data_size=2 -> 4 bytes
  if (pad_en)
    w0 |= (1u << 20) | (pad_interval_code << 22) | (pad_amount_code << 25);
  g1[0] = (int)w0;                                      // mask=0, flags
  g1[1] = (int)((tensor_d0 & 0xFFFFu) << 16);           // tensor_dim0[15:0]
  g1[2] = (int)(((tensor_d0 >> 16) & 0xFFFFu) |
                ((tensor_d1 & 0xFFFFu) << 16));         // d0 hi | d1 lo
  g1[3] = (int)(((tensor_d1 >> 16) & 0xFFFFu) |
                ((tile_d0 & 0xFFFFu) << 16));           // d1 hi | tile_dim0
  g1[4] = (int)(tile_d1 & 0xFFFFu);                     // tile_dim1, tile_dim2=0
  g1[5] = (int)d0_stride;                               // tensor_dim0_stride lo
  g1[6] = 0;                                            // stride hi, d1_stride
  g1[7] = 0;
  v4i g2 = {0, 0, 0, 0};
  v4i g3 = {0, 0, 0, 0};
  v8i g4 = {0, 0, 0, 0, 0, 0, 0, 0};
  __builtin_amdgcn_tensor_load_to_lds(g0, g1, g2, g3, g4, 0);
}
#endif

// ---------------------------------------------------------------------------
// Top-k sparse attention. One workgroup per (bt, head): K rows (stride 66) and
// V rows (stride 64) staged in LDS (~135 KB of 320 KB/WGP) -- via TDM when
// available. Each wave32 handles 32 queries: lanes parallel over the 144
// gathered keys (5 slots/lane), shuffle-reduction softmax, shuffle-broadcast
// PV pass with lanes covering head-dim. Emits bf16 activations.
// ---------------------------------------------------------------------------
__global__ __launch_bounds__(256) void sparse_attn_kernel(
    const float* __restrict__ q, const float* __restrict__ k,
    const float* __restrict__ v, const int* __restrict__ topk,
    const float* __restrict__ out_not, unsigned short* __restrict__ attb) {
  extern __shared__ float smem[];
  float* Ks = smem;                     // [256][66]
  float* Vs = smem + 256 * 66;          // [256][64]
  float* Qs = Vs + 256 * 64;            // [8 waves][64]

  const int bh = blockIdx.x;            // bt*16 + head
  const int bt = bh >> 4, h = bh & 15;
  const float* kb = k + (size_t)bh * S * DHEAD;
  const float* vb = v + (size_t)bh * S * DHEAD;
  const int t = threadIdx.x;
  const int wave = t >> 5, lane = t & 31;

#if ATTN_USE_TDM
  if (wave == 0) {
    // K: 256x64 f32 tile, LDS-padded to stride 66 (64 DW data + 2 DW pad)
    tdm_load_2d_f32(kb, (unsigned)(size_t)(void*)Ks, 64, 256, 64, 256, 64,
                    true, /*interval 64DW*/ 5u, /*amount 2DW*/ 1u);
    // V: 256x64 f32 tile, contiguous
    tdm_load_2d_f32(vb, (unsigned)(size_t)(void*)Vs, 64, 256, 64, 256, 64,
                    false, 0u, 0u);
    __builtin_amdgcn_s_wait_tensorcnt(0);
  }
  __syncthreads();
#else
  for (int r = t; r < 4096; r += 256) {   // 256 rows x 16 float4 chunks
    const int row = r >> 4, c4 = (r & 15) * 4;
    const float4 kv = *(const float4*)&kb[row * 64 + c4];
    float* kd = &Ks[row * 66 + c4];
    kd[0] = kv.x; kd[1] = kv.y; kd[2] = kv.z; kd[3] = kv.w;
    *(float4*)&Vs[row * 64 + c4] = *(const float4*)&vb[row * 64 + c4];
  }
  __syncthreads();
#endif

  const int d0 = lane * 2;

  for (int qi = 0; qi < 32; ++qi) {
    const int s = wave * 32 + qi;
    const float* qrow = q + ((size_t)(bh * S + s)) * DHEAD;
    Qs[wave * 64 + d0]     = qrow[d0];
    Qs[wave * 64 + d0 + 1] = qrow[d0 + 1];
    const int* idxp = topk + ((size_t)(bh * S + s)) * TOPK;

    float sc[5];
    int   idv[5];
#pragma unroll
    for (int jj = 0; jj < 5; ++jj) {
      const int j = jj * 32 + lane;
      float acc = -INFINITY;
      int id = 0;
      if (j < TOPK) {
        id = idxp[j];
        const float2* kr = (const float2*)&Ks[id * 66];
        const float2* q2 = (const float2*)&Qs[wave * 64];
        float a = 0.f;
#pragma unroll
        for (int d = 0; d < 32; ++d) {
          const float2 kk = kr[d], qq = q2[d];
          a += qq.x * kk.x + qq.y * kk.y;
        }
        acc = a * 0.125f;  // DIM_HEAD^-0.5
      }
      sc[jj] = acc;
      idv[jj] = id;
    }

    // wave32 max reduction
    float mx = sc[0];
#pragma unroll
    for (int jj = 1; jj < 5; ++jj) mx = fmaxf(mx, sc[jj]);
#pragma unroll
    for (int off = 16; off; off >>= 1) mx = fmaxf(mx, __shfl_xor(mx, off, 32));

    float ex[5];
    float se = 0.f;
#pragma unroll
    for (int jj = 0; jj < 5; ++jj) {
      ex[jj] = __expf(sc[jj] - mx);  // -inf slots -> 0
      se += ex[jj];
    }
#pragma unroll
    for (int off = 16; off; off >>= 1) se += __shfl_xor(se, off, 32);
    const float inv = 1.f / se;

    // PV pass: broadcast (prob, idx) from owning lane; lanes cover head-dim.
    float o0 = 0.f, o1 = 0.f;
#pragma unroll
    for (int jj = 0; jj < 5; ++jj) {
      const int lim = (jj == 4) ? 16 : 32;  // 144 = 4*32 + 16
      for (int l = 0; l < lim; ++l) {
        const float p  = __shfl(ex[jj], l, 32);
        const int   id = __shfl(idv[jj], l, 32);
        const float2 vv = *(const float2*)&Vs[id * 64 + d0];
        o0 += p * vv.x;
        o1 += p * vv.y;
      }
    }
    o0 *= inv;
    o1 *= inv;

    const float* nb = &out_not[((size_t)(bh * S + s)) * DHEAD + d0];
    o0 += nb[0];
    o1 += nb[1];
    unsigned short* ab = &attb[((size_t)(bt * S + s)) * 1024 + h * 64 + d0];
    ab[0] = f32_to_bf16_rne(o0);
    ab[1] = f32_to_bf16_rne(o1);
  }
}

// ---------------------------------------------------------------------------
// Launch orchestration
// ---------------------------------------------------------------------------
extern "C" void kernel_launch(void* const* d_in, const int* in_sizes, int n_in,
                              void* d_out, int out_size, void* d_ws,
                              size_t ws_size, hipStream_t stream) {
  const float* x       = (const float*)d_in[0];  // (1,2,16,16,1024)
  const float* w_qkv   = (const float*)d_in[1];  // (3072,1024)
  const float* w_out   = (const float*)d_in[2];  // (1024,1024)
  const float* b_out   = (const float*)d_in[3];  // (1024,)
  const float* out_not = (const float*)d_in[4];  // (2,16,256,64)
  const int*   topk    = (const int*)d_in[5];    // (2,16,256,144)

  char* ws = (char*)d_ws;
  size_t off = 0;
  auto alloc = [&](size_t bytes) -> void* {
    void* p = ws + off;
    off = (off + bytes + 255) & ~(size_t)255;
    return p;
  };
  unsigned short* xb   = (unsigned short*)alloc((size_t)512 * 1024 * 2);
  unsigned short* wqb  = (unsigned short*)alloc((size_t)3072 * 1024 * 2);
  unsigned short* wob  = (unsigned short*)alloc((size_t)1024 * 1024 * 2);
  float* qbuf          = (float*)alloc((size_t)BT * HEADS * S * DHEAD * 4);
  float* kbuf          = (float*)alloc((size_t)BT * HEADS * S * DHEAD * 4);
  float* vbuf          = (float*)alloc((size_t)BT * HEADS * S * DHEAD * 4);
  unsigned short* attb = (unsigned short*)alloc((size_t)512 * 1024 * 2);

  // 1) bf16 conversions for WMMA operands
  hipLaunchKernelGGL(f32_to_bf16_kernel, dim3(524288 / 256), dim3(256), 0,
                     stream, x, xb, 524288);
  hipLaunchKernelGGL(f32_to_bf16_kernel, dim3(3145728 / 256), dim3(256), 0,
                     stream, w_qkv, wqb, 3145728);
  hipLaunchKernelGGL(f32_to_bf16_kernel, dim3(1048576 / 256), dim3(256), 0,
                     stream, w_out, wob, 1048576);

  // 2) qkv projection: (512x1024)x(1024x3072) via WMMA, scatter to q/k/v
  hipLaunchKernelGGL((gemm_bf16_wmma<1024, 0>), dim3(512 / BM, 3072 / BN),
                     dim3(256), 0, stream, xb, wqb, (const float*)nullptr,
                     qbuf, kbuf, vbuf);

  // 3) axial RoPE on q and k (in place)
  hipLaunchKernelGGL(rope_kernel, dim3(524288 / 256), dim3(256), 0, stream,
                     qbuf, kbuf);

  // 4) top-k sparse attention, LDS-resident K/V per (bt, head)
  const size_t smem = (size_t)(256 * 66 + 256 * 64 + 8 * 64) * sizeof(float);
  hipLaunchKernelGGL(sparse_attn_kernel, dim3(BT * HEADS), dim3(256), smem,
                     stream, qbuf, kbuf, vbuf, topk, out_not, attb);

  // 5) output projection: (512x1024)x(1024x1024) + bias via WMMA
  hipLaunchKernelGGL((gemm_bf16_wmma<1024, 1>), dim3(512 / BM, 1024 / BN),
                     dim3(256), 0, stream, attb, wob, b_out, (float*)d_out,
                     (float*)nullptr, (float*)nullptr);
}
